// Pedal_26482768347627
// MI455X (gfx1250) — matrix operand order repaired and checked by reading, not compiled, not run on hardware
//
#include <hip/hip_runtime.h>
#include <hip/hip_bf16.h>

#define B_DIM 128
#define D_DIM 128
#define P_DIM 4
#define M_TOT 100000
#define K_POS 10
#define SCALE_F 10.0f

typedef __attribute__((ext_vector_type(8)))  float  v8f;
typedef __attribute__((ext_vector_type(16))) __bf16 v16bf;
typedef __attribute__((ext_vector_type(8)))  __bf16 v8bf;
typedef __attribute__((ext_vector_type(4)))  __bf16 v4bf;

union V16U { v16bf v; v8bf h[2]; };

// ---- CDNA5 feature availability (device pass only; host pass sees 0) ----
#ifndef __has_builtin
#define __has_builtin(x) 0
#endif
#if __has_builtin(__builtin_amdgcn_tensor_load_to_lds)
#define HAVE_TDM 1
#else
#define HAVE_TDM 0
#endif
#if __has_builtin(__builtin_amdgcn_s_wait_tensorcnt)
#define WAIT_TENSORCNT(n) __builtin_amdgcn_s_wait_tensorcnt(n)
#else
#define WAIT_TENSORCNT(n) asm volatile("s_wait_tensorcnt " #n ::: "memory")
#endif

// fast sqrt: raw v_sqrt_f32 (no IEEE fixup sequence); feeds exp() so ~2ulp ok
#if __has_builtin(__builtin_amdgcn_sqrtf)
#define FAST_SQRT(x) __builtin_amdgcn_sqrtf(x)
#else
#define FAST_SQRT(x) __builtin_sqrtf(x)
#endif

#if HAVE_TDM
typedef __attribute__((ext_vector_type(4))) unsigned int v4u;
typedef __attribute__((ext_vector_type(4))) int          v4i;
typedef __attribute__((ext_vector_type(8))) int          v8i;

// DMA a 64-row x 128-col f32 tile (row stride 128 f32) from global into LDS.
// D# built per CDNA5 ISA ch.8: group0 = {count,lds_addr,global_addr,type=2},
// group1 = {data_size=4B, tensor_dim0=128, tensor_dim1=64, tile 128x64,
//           dim0_stride=128}; groups 2/3 zero (2D tensor).
__device__ inline void tdm_load_tile64x128(const float* gsrc, float* lds_dst) {
  unsigned long long ga = (unsigned long long)(size_t)gsrc;
  unsigned int lo = (unsigned int)(size_t)(void*)lds_dst;  // LDS byte offset
  v4u g0;
  g0[0] = 1u;                                        // count=1 (valid user D#)
  g0[1] = lo;                                        // lds_addr
  g0[2] = (unsigned int)(ga & 0xFFFFFFFFu);          // global_addr[31:0]
  g0[3] = (unsigned int)((ga >> 32) & 0x01FFFFFFu)   // global_addr[56:32]
          | (2u << 30);                              // type=2 ("image")
  v8i g1;
  g1[0] = (int)(2u << 16);    // workgroup_mask=0, data_size=2 -> 4 bytes
  g1[1] = (int)(128u << 16);  // tensor_dim0[15:0]=128 @bits 63:48
  g1[2] = (int)(64u << 16);   // tensor_dim0 hi=0; tensor_dim1[15:0]=64 @95:80
  g1[3] = (int)(128u << 16);  // tensor_dim1 hi=0; tile_dim0=128 @127:112
  g1[4] = (int)64u;           // tile_dim1=64; tile_dim2=0
  g1[5] = (int)128u;          // tensor_dim0_stride[31:0]=128
  g1[6] = 0;                  // stride hi, tensor_dim1_stride lo
  g1[7] = 0;
  v4i gz = {0, 0, 0, 0};
#if __clang_major__ >= 23
  v8i gz8 = {0, 0, 0, 0, 0, 0, 0, 0};
  __builtin_amdgcn_tensor_load_to_lds(g0, g1, gz, gz, gz8, 0);
#else
  __builtin_amdgcn_tensor_load_to_lds(g0, g1, gz, gz, 0);
#endif
}
#endif  // HAVE_TDM

// ---------------------------------------------------------------- init
__global__ void mc_init(float* __restrict__ mask, float* __restrict__ negsum,
                        float* __restrict__ acc) {
  int i = blockIdx.x * blockDim.x + threadIdx.x;
  if (i < M_TOT) mask[i] = 1.0f;
  if (i < P_DIM * B_DIM) negsum[i] = 0.0f;
  if (i < 8) acc[i] = 0.0f;
}

// ------------------------------------------------- mask scatter + pos_vid
__global__ void mc_scatter(const int* __restrict__ cidx, const int* __restrict__ pos,
                           const int* __restrict__ mvid, float* __restrict__ mask,
                           float* __restrict__ out) {
  int i = blockIdx.x * blockDim.x + threadIdx.x;
  if (i < B_DIM * K_POS) {
    int m = cidx[i];
    mask[m] = 0.0f;
    out[3 + i] = (float)mvid[m];
  }
  if (i < B_DIM) mask[pos[i]] = 0.0f;
}

// ---------------------------------------------------------- row normalize
__global__ __launch_bounds__(128) void mc_norm(const float* __restrict__ gimg,
                                               const float* __restrict__ gtxt,
                                               float* __restrict__ img_n,
                                               float* __restrict__ txt_n) {
  int row = blockIdx.x;
  const float* src = blockIdx.y ? gtxt : gimg;
  float* dst = blockIdx.y ? txt_n : img_n;
  int t = threadIdx.x;
  float v = src[row * D_DIM + t];
  float ss = v * v;
#pragma unroll
  for (int o = 16; o >= 1; o >>= 1) ss += __shfl_xor(ss, o, 32);
  __shared__ float part[4];
  if ((t & 31) == 0) part[t >> 5] = ss;
  __syncthreads();
  float tot = part[0] + part[1] + part[2] + part[3];
  dst[row * D_DIM + t] = v * rsqrtf(tot);
}

// ------------------------------------------------------------- MMD pairs
__global__ __launch_bounds__(256) void mc_mmd(const float* __restrict__ img_n,
                                              const float* __restrict__ txt_n,
                                              float* __restrict__ acc) {
  int gid = blockIdx.x * 256 + threadIdx.x;
  int i = gid >> 7, j = gid & 127;
  const float* ai = img_n + (long)i * D_DIM;
  const float* aj = img_n + (long)j * D_DIM;
  const float* ti = txt_n + (long)i * D_DIM;
  const float* tj = txt_n + (long)j * D_DIM;
  float di = 0.f, dt = 0.f, dc = 0.f;
  for (int d = 0; d < D_DIM; d++) {
    di += ai[d] * aj[d];
    dt += ti[d] * tj[d];
    dc += ai[d] * tj[d];
  }
  float kxx = __expf(-fmaxf(2.f - 2.f * di, 0.f) * 0.5f);
  float kyy = __expf(-fmaxf(2.f - 2.f * dt, 0.f) * 0.5f);
  float kxy = __expf(-fmaxf(2.f - 2.f * dc, 0.f) * 0.5f);
  float a0 = (i != j) ? kxx : 0.f;
  float a1 = (i != j) ? kyy : 0.f;
  __shared__ float s3[3];
  if (threadIdx.x < 3) s3[threadIdx.x] = 0.f;
  __syncthreads();
  atomicAdd(&s3[0], a0);
  atomicAdd(&s3[1], a1);
  atomicAdd(&s3[2], kxy);
  __syncthreads();
  if (threadIdx.x < 3) atomicAdd(&acc[threadIdx.x], s3[threadIdx.x]);
}

// ------------------------------------------- main WMMA neg-distance kernel
// grid = (ceil(M/512), P), block = 256 (8 wave32s).
// Block covers 512 centers in 8 sub-tiles of 64. TDM double-buffers the next
// f32 sub-tile into dynamic LDS while the current one is converted/consumed.
__global__ __launch_bounds__(256) void mc_neg(const float* __restrict__ lfeat,
                                              const float* __restrict__ centers,
                                              const float* __restrict__ mask,
                                              float* __restrict__ negsum) {
  constexpr int AST = 136;  // bf16 row stride (272B: 16B aligned, conflict-free)
  __shared__ __align__(16) __bf16 aT[B_DIM * AST];  // 34,816 B
  __shared__ __align__(16) __bf16 cT[64 * AST];     // 17,408 B
  __shared__ float c2s[64];
  __shared__ float f2s[B_DIM];
  __shared__ float wms[64];
  extern __shared__ float stg[];  // 2 x (64*128) f32 TDM staging (65,536 B)

  const int tid = threadIdx.x;
  const int lane = tid & 31;
  const int wave = tid >> 5;
  const int p = blockIdx.y;
  const long mBase0 = (long)blockIdx.x * 512;
  const float* cbase = centers + (long)p * M_TOT * D_DIM;

  if (tid < B_DIM) f2s[tid] = 0.0f;
  __syncthreads();

  // ---- stage A = local_img_feats[p] (128x128 f32 -> bf16), compute |f|^2
  {
    int row = tid >> 1;
    int colb = (tid & 1) * 64;
    const float* sp = lfeat + (long)p * B_DIM * D_DIM + (long)row * D_DIM + colb;
    __bf16* dp = aT + row * AST + colb;
    float ss = 0.0f;
#pragma unroll
    for (int c = 0; c < 64; c += 4) {
      float4 v = *(const float4*)(sp + c);
      ss += v.x * v.x + v.y * v.y + v.z * v.z + v.w * v.w;
      v4bf o;
      o[0] = (__bf16)v.x; o[1] = (__bf16)v.y; o[2] = (__bf16)v.z; o[3] = (__bf16)v.w;
      *(v4bf*)(dp + c) = o;
    }
    atomicAdd(&f2s[row], ss);
  }

#if HAVE_TDM
  // prologue: kick off DMA of sub-tile 0 (wave 0 only; EXEC-independent op)
  if (wave == 0) {
    if (mBase0 + 64 <= M_TOT)
      tdm_load_tile64x128(cbase + mBase0 * D_DIM, stg);
  }
#endif
  __syncthreads();

  // ---- A fragments for this wave's 16-row stripe (16-bit A 16x32 layout)
  v16bf afrag[4];
  {
    int r = wave * 16 + (lane & 15);
    int khalf = (lane >> 4) * 8;
#pragma unroll
    for (int kc = 0; kc < 4; kc++) {
      V16U t;
      t.h[0] = *(const v8bf*)(aT + r * AST + kc * 32 + khalf);
      t.h[1] = *(const v8bf*)(aT + r * AST + kc * 32 + khalf + 16);
      afrag[kc] = t.v;
    }
  }

  const int rbase = wave * 16 + (lane >> 4) * 8;  // C rows per half-wave
  // hoist loop-invariant row norms |f|^2 into registers (f2s is final here)
  float f2r[8];
#pragma unroll
  for (int e = 0; e < 8; e++) f2r[e] = f2s[rbase + e];

  float accv[8];
#pragma unroll
  for (int e = 0; e < 8; e++) accv[e] = 0.0f;

  for (int s = 0; s < 8; s++) {
    long mBase = mBase0 + (long)s * 64;
    const bool fullCur = (mBase + 64 <= M_TOT);
    __syncthreads();              // previous sub-tile compute done
    if (tid < 64) c2s[tid] = 0.0f;
#if HAVE_TDM
    if (wave == 0) {
      long mNext = mBase + 64;
      bool fullNext = (s < 7) && (mNext + 64 <= M_TOT);
      if (fullNext)               // DMA s+1 into the other buffer
        tdm_load_tile64x128(cbase + mNext * D_DIM, stg + ((s + 1) & 1) * (64 * 128));
      if (fullCur) {              // wait for the *older* (current) DMA only
        if (fullNext) { WAIT_TENSORCNT(1); } else { WAIT_TENSORCNT(0); }
      }
    }
#endif
    __syncthreads();              // c2 zeroed AND current staging ready

    // ---- convert 64 centers rows to bf16 cT, |c|^2, stage mask row
    {
      int j = tid >> 2;
      int q = tid & 3;
      int colb = q * 32;
      long m = mBase + j;
      __bf16* dp = cT + j * AST + colb;
      float ss = 0.0f;
      bool useLds = false;
#if HAVE_TDM
      useLds = fullCur;
#endif
      if (useLds) {
#if HAVE_TDM
        const float* sp = stg + (s & 1) * (64 * 128) + j * 128 + colb;
#pragma unroll
        for (int c = 0; c < 32; c += 4) {
          float4 v = *(const float4*)(sp + c);
          ss += v.x * v.x + v.y * v.y + v.z * v.z + v.w * v.w;
          v4bf o;
          o[0] = (__bf16)v.x; o[1] = (__bf16)v.y; o[2] = (__bf16)v.z; o[3] = (__bf16)v.w;
          *(v4bf*)(dp + c) = o;
        }
#endif
      } else if (m < M_TOT) {
        const float* sp = cbase + m * D_DIM + colb;
        if (m + 64 < M_TOT) __builtin_prefetch(sp + 64 * D_DIM, 0, 0);
#pragma unroll
        for (int c = 0; c < 32; c += 4) {
          float4 v = *(const float4*)(sp + c);
          ss += v.x * v.x + v.y * v.y + v.z * v.z + v.w * v.w;
          v4bf o;
          o[0] = (__bf16)v.x; o[1] = (__bf16)v.y; o[2] = (__bf16)v.z; o[3] = (__bf16)v.w;
          *(v4bf*)(dp + c) = o;
        }
      } else {
        v4bf z;
        z[0] = (__bf16)0.f; z[1] = (__bf16)0.f; z[2] = (__bf16)0.f; z[3] = (__bf16)0.f;
#pragma unroll
        for (int c = 0; c < 32; c += 4) *(v4bf*)(dp + c) = z;
      }
      atomicAdd(&c2s[j], ss);
      if (q == 0) wms[j] = (m < M_TOT) ? mask[m] : 0.0f;  // mask row to LDS
    }
    __syncthreads();

    // ---- 4 col-tiles of 16 centers; K=128 in 4 WMMA chunks each
#pragma unroll
    for (int ct = 0; ct < 4; ct++) {
      int n = ct * 16 + (lane & 15);      // center column for this lane
      int kh = (lane >> 4) * 8;           // B 32x16: K 8h..8h+7 and 16+8h..16+8h+7 per lane
      v8f cacc = {0.f, 0.f, 0.f, 0.f, 0.f, 0.f, 0.f, 0.f};
#pragma unroll
      for (int kc = 0; kc < 4; kc++) {
        V16U t;
        t.h[0] = *(const v8bf*)(cT + n * AST + kc * 32 + kh);
        t.h[1] = *(const v8bf*)(cT + n * AST + kc * 32 + kh + 16);
        cacc = __builtin_amdgcn_wmma_f32_16x16x32_bf16(
            false, afrag[kc], false, t.v, (short)0, cacc, false, false);
      }
      float wm = wms[n];
      float c2v = c2s[n];
#pragma unroll
      for (int e = 0; e < 8; e++) {
        float d2 = fmaxf(f2r[e] + c2v - 2.0f * cacc[e], 0.0f);
        accv[e] += wm * __expf(-SCALE_F * FAST_SQRT(d2));
      }
    }
  }

  // ---- reduce across the 16 column-lanes of each half-wave
#pragma unroll
  for (int off = 8; off >= 1; off >>= 1)
#pragma unroll
    for (int e = 0; e < 8; e++) accv[e] += __shfl_xor(accv[e], off, 32);

  if ((lane & 15) == 0) {
#pragma unroll
    for (int e = 0; e < 8; e++)
      atomicAdd(&negsum[p * B_DIM + rbase + e], accv[e]);
  }
}

// ------------------------------------------- positive term + per-p loss
__global__ __launch_bounds__(128) void mc_pos(const float* __restrict__ lfeat,
                                              const float* __restrict__ centers,
                                              const int* __restrict__ cidx,
                                              const float* __restrict__ negsum,
                                              float* __restrict__ acc) {
  int p = blockIdx.x, b = threadIdx.x;
  const float* f = lfeat + ((long)p * B_DIM + b) * D_DIM;
  float f2 = 0.f;
  for (int d = 0; d < D_DIM; d++) f2 += f[d] * f[d];
  float sum = 0.f;
  for (int k = 0; k < K_POS; k++) {
    int idx = cidx[b * K_POS + k];
    const float* c = centers + ((long)p * M_TOT + idx) * D_DIM;
    float p2 = 0.f, fp = 0.f;
    for (int d = 0; d < D_DIM; d++) {
      p2 += c[d] * c[d];
      fp += f[d] * c[d];
    }
    float d2 = fmaxf(f2 + p2 - 2.f * fp, 0.f);
    sum += __expf(-SCALE_F * FAST_SQRT(d2));
  }
  float x = __logf(sum);
  float y = __logf(negsum[p * B_DIM + b]);
  float ss = -x + y;
#pragma unroll
  for (int o = 16; o >= 1; o >>= 1) ss += __shfl_xor(ss, o, 32);
  __shared__ float part[4];
  if ((b & 31) == 0) part[b >> 5] = ss;
  __syncthreads();
  if (b == 0) {
    float l = (part[0] + part[1] + part[2] + part[3]) / (float)B_DIM;
    if (l != l) l = 0.f;  // nan guard per reference
    atomicAdd(&acc[3], l);
  }
}

// ---------------------------------------------------------------- finalize
__global__ void mc_fin(const float* __restrict__ acc, float* __restrict__ out) {
  if (threadIdx.x == 0 && blockIdx.x == 0) {
    float denom = (float)B_DIM * (float)(B_DIM - 1);
    float mmd = acc[0] / denom + acc[1] / denom -
                2.0f * acc[2] / ((float)B_DIM * (float)B_DIM);
    float local = acc[3] / (float)P_DIM;
    out[0] = local + mmd;
    out[1] = local;
    out[2] = mmd;
  }
}

extern "C" void kernel_launch(void* const* d_in, const int* in_sizes, int n_in,
                              void* d_out, int out_size, void* d_ws, size_t ws_size,
                              hipStream_t stream) {
  const float* gimg = (const float*)d_in[0];
  const float* gtxt = (const float*)d_in[1];
  const float* lfeat = (const float*)d_in[2];
  const float* ctr = (const float*)d_in[3];
  const int* cidx = (const int*)d_in[4];
  const int* pos = (const int*)d_in[5];
  const int* mvid = (const int*)d_in[6];
  float* out = (float*)d_out;

  // workspace layout (floats)
  float* ws = (float*)d_ws;
  float* mask = ws;                        // M_TOT
  float* negsum = mask + M_TOT;            // P*B
  float* img_n = negsum + P_DIM * B_DIM;   // B*D
  float* txt_n = img_n + B_DIM * D_DIM;    // B*D
  float* acc = txt_n + B_DIM * D_DIM;      // 8 scalars

  // allow 64KB dynamic LDS for the TDM staging buffers (idempotent; not a
  // stream op, safe under graph capture)
  (void)hipFuncSetAttribute((const void*)mc_neg,
                            hipFuncAttributeMaxDynamicSharedMemorySize, 65536);

  mc_init<<<(M_TOT + 255) / 256, 256, 0, stream>>>(mask, negsum, acc);
  mc_scatter<<<(B_DIM * K_POS + 255) / 256, 256, 0, stream>>>(cidx, pos, mvid, mask, out);
  mc_norm<<<dim3(B_DIM, 2), 128, 0, stream>>>(gimg, gtxt, img_n, txt_n);
  mc_mmd<<<(B_DIM * B_DIM) / 256, 256, 0, stream>>>(img_n, txt_n, acc);
  mc_neg<<<dim3((M_TOT + 511) / 512, P_DIM), 256, 65536, stream>>>(lfeat, ctr, mask, negsum);
  mc_pos<<<P_DIM, 128, 0, stream>>>(lfeat, ctr, cidx, negsum, acc);
  mc_fin<<<1, 1, 0, stream>>>(acc, out);
}
